// DecoupledPointJAFAR_10462540333134
// MI455X (gfx1250) — matrix operand order repaired and checked by reading, not compiled, not run on hardware
//
#include <hip/hip_runtime.h>
#include <hip/hip_bf16.h>
#include <math.h>

// ---------------------------------------------------------------------------
// DecoupledPointJAFAR on MI455X (gfx1250, wave32, WMMA).
//
// Sizes: B=4, N=32768, BN=131072, QK=64, KNN=16, GEO=12, SEM=192, NC=13.
//
//  * All 1x1 convs -> templated WMMA f16 GEMM (16x16x32 tiles); compile-time
//    shapes give unguarded float4 loads (global_load_b128) and branch-free
//    stores (full row tiles; only the 13-wide cls head keeps a col guard).
//  * Irregular operands padded once: jafar 12->32 cols, cls_w 13->16 rows.
//  * GroupNorm -> two-pass (LDS-reduced stats, then apply).
//  * pos_enc never materialized:  Q . pos_enc = (W2^T Q) . h + Q . b2, with
//    GroupNorm stats of h = W1 rel + b1 derived analytically from per-batch
//    1st/2nd moments of rel_pos.
//  * Fused attention: one wave32 per point, 2 channels/lane, __shfl_xor dots.
// ---------------------------------------------------------------------------

typedef __attribute__((ext_vector_type(16))) _Float16 v16h;
typedef __attribute__((ext_vector_type(8)))  float    v8f;

#define BB   4
#define NN   32768
#define BN   131072
#define QK   64
#define KNN  16
#define GEO  12
#define SEM  192
#define NC   13

// -------------------- templated WMMA GEMM:  Y = X * W^T + b -----------------
// X: [Nrows, C] f32 (C % 32 == 0), W: [O, C] f32 (O % 16 == 0),
// Y: [Nrows, OST] f32. Requires Nrows % 64 == 0 (full tiles, no row guard).
// Block = 128 threads = 4 waves; each wave owns 16 rows.
template <int C, int O, int OST>
__global__ __launch_bounds__(128) void gemm_t(const float* __restrict__ X,
                                              const float* __restrict__ W,
                                              const float* __restrict__ bias,
                                              float* __restrict__ Y, int Nrows) {
  constexpr int nKT = C / 32;
  constexpr int nOT = O / 16;
  const int lane  = threadIdx.x & 31;
  const int wave  = threadIdx.x >> 5;
  const int rowBase = (blockIdx.x * 4 + wave) * 16;
  if (rowBase >= Nrows) return;
  const int m     = lane & 15;   // A: row in tile; B/D: col in tile
  const int khalf = lane >> 4;
  const int row   = rowBase + m;

  // ---- load all A fragments for this row (unguarded, vectorized) ----
  // Layout: lanes 0-15 hold row m, K {0..7, 16..23}; lanes 16-31: K {8..15, 24..31}.
  v16h a[nKT];
  const float* xrow = X + (size_t)row * C;
  #pragma unroll
  for (int kt = 0; kt < nKT; ++kt) {
    const float* p = xrow + kt * 32 + khalf * 8;
    const float4 xa = *(const float4*)(p + 0);
    const float4 xb = *(const float4*)(p + 4);
    const float4 xc = *(const float4*)(p + 16);
    const float4 xd = *(const float4*)(p + 20);
    a[kt][0]  = (_Float16)xa.x; a[kt][1]  = (_Float16)xa.y;
    a[kt][2]  = (_Float16)xa.z; a[kt][3]  = (_Float16)xa.w;
    a[kt][4]  = (_Float16)xb.x; a[kt][5]  = (_Float16)xb.y;
    a[kt][6]  = (_Float16)xb.z; a[kt][7]  = (_Float16)xb.w;
    a[kt][8]  = (_Float16)xc.x; a[kt][9]  = (_Float16)xc.y;
    a[kt][10] = (_Float16)xc.z; a[kt][11] = (_Float16)xc.w;
    a[kt][12] = (_Float16)xd.x; a[kt][13] = (_Float16)xd.y;
    a[kt][14] = (_Float16)xd.z; a[kt][15] = (_Float16)xd.w;
  }

  #pragma unroll
  for (int ot = 0; ot < nOT; ++ot) {
    const int col = ot * 16 + m;
    v8f acc = {};
    #pragma unroll
    for (int kt = 0; kt < nKT; ++kt) {
      // B layout: lane holds column `col`; lanes 0-15: K {0..7, 16..23}, lanes 16-31: K {8..15, 24..31}.
      const float* p = W + (size_t)col * C + kt * 32 + khalf * 8;
      const float4 wa = *(const float4*)(p + 0);
      const float4 wb = *(const float4*)(p + 4);
      const float4 wc = *(const float4*)(p + 16);
      const float4 wd = *(const float4*)(p + 20);
      v16h bm;
      bm[0]  = (_Float16)wa.x; bm[1]  = (_Float16)wa.y;
      bm[2]  = (_Float16)wa.z; bm[3]  = (_Float16)wa.w;
      bm[4]  = (_Float16)wb.x; bm[5]  = (_Float16)wb.y;
      bm[6]  = (_Float16)wb.z; bm[7]  = (_Float16)wb.w;
      bm[8]  = (_Float16)wc.x; bm[9]  = (_Float16)wc.y;
      bm[10] = (_Float16)wc.z; bm[11] = (_Float16)wc.w;
      bm[12] = (_Float16)wd.x; bm[13] = (_Float16)wd.y;
      bm[14] = (_Float16)wd.z; bm[15] = (_Float16)wd.w;
      acc = __builtin_amdgcn_wmma_f32_16x16x32_f16(
          false, a[kt], false, bm, (short)0, acc, false, false);
    }
    // Store: D layout lane l -> col = ot*16 + (l&15), rows rowBase + r + 8*(l>>4).
    // Full row tiles (Nrows % 64 == 0): no row guard. Col guard only if OST < O.
    if (OST == O || col < OST) {
      const float bv = bias ? bias[col] : 0.f;
      float* yp = Y + (size_t)(rowBase + khalf * 8) * OST + col;
      #pragma unroll
      for (int r = 0; r < 8; ++r) yp[(size_t)r * OST] = acc[r] + bv;
    }
  }
}

// -------------------- padding helpers ---------------------------------------
__global__ void pad_jafar(const float* __restrict__ in, float* __restrict__ out) {
  const int i = blockIdx.x * blockDim.x + threadIdx.x;
  if (i >= BN * 32) return;
  const int r = i >> 5, c = i & 31;
  out[i] = (c < GEO) ? in[(size_t)r * GEO + c] : 0.f;
}

__global__ void pad_clsw(const float* __restrict__ in, float* __restrict__ out) {
  const int i = blockIdx.x * blockDim.x + threadIdx.x;
  if (i >= 16 * 64) return;
  const int r = i >> 6;
  out[i] = (r < NC) ? in[i] : 0.f;
}

// -------------------- GroupNorm: stats + apply ------------------------------
__global__ void gn_stats(const float* __restrict__ X, float* __restrict__ stats,
                         int Npts, int C, int groups) {
  const int b  = blockIdx.x / groups;
  const int g  = blockIdx.x % groups;
  const int gs = C / groups;
  const int c0 = g * gs;
  float s = 0.f, s2 = 0.f;
  for (int n = threadIdx.x; n < Npts; n += blockDim.x) {
    const float* p = X + ((size_t)b * Npts + n) * C + c0;
    for (int c = 0; c < gs; ++c) { const float v = p[c]; s += v; s2 += v * v; }
  }
  __shared__ float sh[256], sh2[256];
  sh[threadIdx.x] = s; sh2[threadIdx.x] = s2;
  __syncthreads();
  for (int st = 128; st > 0; st >>= 1) {
    if (threadIdx.x < st) {
      sh[threadIdx.x]  += sh[threadIdx.x + st];
      sh2[threadIdx.x] += sh2[threadIdx.x + st];
    }
    __syncthreads();
  }
  if (threadIdx.x == 0) {
    const float M = (float)Npts * (float)gs;
    const float mean = sh[0] / M;
    const float var  = sh2[0] / M - mean * mean;
    stats[blockIdx.x * 2 + 0] = mean;
    stats[blockIdx.x * 2 + 1] = rsqrtf(var + 1e-5f);
  }
}

__global__ void gn_apply(float* __restrict__ X, const float* __restrict__ stats,
                         const float* __restrict__ gw, const float* __restrict__ gb,
                         int total, int C, int groups, int Npts, int relu) {
  const int i = blockIdx.x * blockDim.x + threadIdx.x;
  if (i >= total) return;
  const int row = i / C, c = i % C;
  const int b = row / Npts;
  const int g = c / (C / groups);
  const float mean = stats[(b * groups + g) * 2 + 0];
  const float rstd = stats[(b * groups + g) * 2 + 1];
  float v = (X[i] - mean) * rstd * gw[c] + gb[c];
  if (relu) v = fmaxf(v, 0.f);
  X[i] = v;
}

// -------------------- boundary head 2nd conv (32 -> 1) ----------------------
__global__ void bdy_head2(const float* __restrict__ H, const float* __restrict__ w,
                          const float* __restrict__ b, float* __restrict__ out) {
  const int pt = blockIdx.x * blockDim.x + threadIdx.x;
  if (pt >= BN) return;
  float s = b[0];
  const float* h = H + (size_t)pt * 32;
  #pragma unroll
  for (int c = 0; c < 32; ++c) s = fmaf(h[c], w[c], s);
  out[pt] = s;
}

// -------------------- small helpers -----------------------------------------
__global__ void transpose64(const float* __restrict__ W, float* __restrict__ WT) {
  const int i = blockIdx.x * blockDim.x + threadIdx.x;
  if (i >= 64 * 64) return;
  const int r = i >> 6, c = i & 63;
  WT[c * 64 + r] = W[i];
}

__global__ void zero_f32(float* __restrict__ p, int n) {
  const int i = blockIdx.x * blockDim.x + threadIdx.x;
  if (i < n) p[i] = 0.f;
}

// -------------------- rel_pos moments (per batch) ---------------------------
// mom[b*9 + {0..2}] = sum rel_j ; mom[b*9 + {3..8}] = sum rel_i*rel_j (00,01,02,11,12,22)
__global__ void rel_moments(const float* __restrict__ xyz, const int* __restrict__ knn,
                            float* __restrict__ mom) {
  const int b = blockIdx.y;
  float a[9];
  #pragma unroll
  for (int j = 0; j < 9; ++j) a[j] = 0.f;
  const int total  = NN * KNN;
  const int stride = gridDim.x * blockDim.x;
  for (int t = blockIdx.x * blockDim.x + threadIdx.x; t < total; t += stride) {
    const int n = t / KNN, k = t % KNN;
    const int idx = knn[((size_t)b * NN + n) * KNN + k];
    const float* pn = xyz + ((size_t)b * NN + n) * 3;
    const float* pg = xyz + ((size_t)b * NN + idx) * 3;
    const float r0 = pn[0] - pg[0], r1 = pn[1] - pg[1], r2 = pn[2] - pg[2];
    a[0] += r0; a[1] += r1; a[2] += r2;
    a[3] += r0 * r0; a[4] += r0 * r1; a[5] += r0 * r2;
    a[6] += r1 * r1; a[7] += r1 * r2; a[8] += r2 * r2;
  }
  __shared__ float sh[256];
  for (int j = 0; j < 9; ++j) {
    sh[threadIdx.x] = a[j];
    __syncthreads();
    for (int st = 128; st > 0; st >>= 1) {
      if (threadIdx.x < st) sh[threadIdx.x] += sh[threadIdx.x + st];
      __syncthreads();
    }
    if (threadIdx.x == 0) atomicAdd(&mom[b * 9 + j], sh[0]);
    __syncthreads();
  }
}

// Analytic GroupNorm stats of h = W1*rel + b1 over (8ch x N x K) per (b,group).
__global__ void rp_finalize(const float* __restrict__ mom, const float* __restrict__ w1,
                            const float* __restrict__ b1, float* __restrict__ rpstats) {
  const int tid = threadIdx.x;
  if (tid >= BB * 8) return;
  const int b = tid / 8, g = tid % 8;
  const float M = (float)NN * (float)KNN;
  const float e0 = mom[b * 9 + 0] / M, e1 = mom[b * 9 + 1] / M, e2m = mom[b * 9 + 2] / M;
  const float s00 = mom[b * 9 + 3] / M, s01 = mom[b * 9 + 4] / M, s02 = mom[b * 9 + 5] / M;
  const float s11 = mom[b * 9 + 6] / M, s12 = mom[b * 9 + 7] / M, s22 = mom[b * 9 + 8] / M;
  float sm = 0.f, se2 = 0.f;
  for (int c = g * 8; c < g * 8 + 8; ++c) {
    const float a0 = w1[c * 3 + 0], a1 = w1[c * 3 + 1], a2 = w1[c * 3 + 2], bb = b1[c];
    const float lin = a0 * e0 + a1 * e1 + a2 * e2m;
    const float quad = a0 * a0 * s00 + a1 * a1 * s11 + a2 * a2 * s22
                     + 2.f * (a0 * a1 * s01 + a0 * a2 * s02 + a1 * a2 * s12);
    sm  += lin + bb;
    se2 += quad + 2.f * bb * lin + bb * bb;
  }
  sm *= 0.125f; se2 *= 0.125f;
  const float var = se2 - sm * sm;
  rpstats[tid * 2 + 0] = sm;
  rpstats[tid * 2 + 1] = rsqrtf(var + 1e-5f);
}

// -------------------- fused attention ---------------------------------------
__device__ __forceinline__ float wred32(float v) {
  #pragma unroll
  for (int off = 16; off >= 1; off >>= 1) v += __shfl_xor(v, off, 32);
  return v;
}

__global__ void attn_fused(const float* __restrict__ Q,  const float* __restrict__ Qw,
                           const float* __restrict__ Kf, const float* __restrict__ V,
                           const float* __restrict__ xyz, const int* __restrict__ knn,
                           const float* __restrict__ rpw1, const float* __restrict__ rpb1,
                           const float* __restrict__ rpgnw, const float* __restrict__ rpgnb,
                           const float* __restrict__ rpstats, const float* __restrict__ b2,
                           float* __restrict__ out_aff, float* __restrict__ out_refined,
                           int* __restrict__ out_knn) {
  const int wave = threadIdx.x >> 5, lane = threadIdx.x & 31;
  const int pt = blockIdx.x * (blockDim.x >> 5) + wave;
  if (pt >= BN) return;
  const int b = pt >> 15;               // N = 32768
  const int c0 = lane, c1 = lane + 32;

  const float q0  = Q[(size_t)pt * QK + c0],  q1  = Q[(size_t)pt * QK + c1];
  const float qw0 = Qw[(size_t)pt * QK + c0], qw1 = Qw[(size_t)pt * QK + c1];
  const float xn0 = xyz[(size_t)pt * 3 + 0];
  const float xn1 = xyz[(size_t)pt * 3 + 1];
  const float xn2 = xyz[(size_t)pt * 3 + 2];

  const float w00 = rpw1[c0 * 3], w01 = rpw1[c0 * 3 + 1], w02 = rpw1[c0 * 3 + 2], bb0 = rpb1[c0];
  const float w10 = rpw1[c1 * 3], w11 = rpw1[c1 * 3 + 1], w12 = rpw1[c1 * 3 + 2], bb1 = rpb1[c1];
  const int   g0 = c0 >> 3, g1 = c1 >> 3;
  const float m0 = rpstats[(b * 8 + g0) * 2], rs0 = rpstats[(b * 8 + g0) * 2 + 1];
  const float m1 = rpstats[(b * 8 + g1) * 2], rs1 = rpstats[(b * 8 + g1) * 2 + 1];
  const float gw0 = rpgnw[c0], gb0 = rpgnb[c0];
  const float gw1 = rpgnw[c1], gb1 = rpgnb[c1];

  const float qb2 = wred32(q0 * b2[c0] + q1 * b2[c1]);

  float logit[KNN];
  int   nb[KNN];
  #pragma unroll
  for (int k = 0; k < KNN; ++k) {
    const int idx = knn[(size_t)pt * KNN + k];
    nb[k] = idx;
    const size_t gp = (size_t)b * NN + idx;
    const float r0 = xn0 - xyz[gp * 3 + 0];
    const float r1 = xn1 - xyz[gp * 3 + 1];
    const float r2 = xn2 - xyz[gp * 3 + 2];
    float h0 = fmaf(w00, r0, fmaf(w01, r1, fmaf(w02, r2, bb0)));
    float h1 = fmaf(w10, r0, fmaf(w11, r1, fmaf(w12, r2, bb1)));
    h0 = fmaxf((h0 - m0) * rs0 * gw0 + gb0, 0.f);
    h1 = fmaxf((h1 - m1) * rs1 * gw1 + gb1, 0.f);
    const float part = qw0 * h0 + qw1 * h1
                     + q0 * Kf[gp * QK + c0] + q1 * Kf[gp * QK + c1];
    logit[k] = (wred32(part) + qb2) * 0.125f;   // 1/sqrt(64)
  }

  float mx = logit[0];
  #pragma unroll
  for (int k = 1; k < KNN; ++k) mx = fmaxf(mx, logit[k]);
  float se = 0.f;
  #pragma unroll
  for (int k = 0; k < KNN; ++k) { logit[k] = __expf(logit[k] - mx); se += logit[k]; }
  const float rse = 1.f / se;

  float r0acc = V[(size_t)pt * QK + c0];
  float r1acc = V[(size_t)pt * QK + c1];
  #pragma unroll
  for (int k = 0; k < KNN; ++k) {
    const float aw = logit[k] * rse;
    const size_t gp = (size_t)b * NN + nb[k];
    r0acc = fmaf(aw, V[gp * QK + c0], r0acc);
    r1acc = fmaf(aw, V[gp * QK + c1], r1acc);
  }
  if (lane < KNN) {
    out_aff[(size_t)pt * KNN + lane] = logit[lane] * rse;
    out_knn[(size_t)pt * KNN + lane] = nb[lane];
  }
  out_refined[(size_t)pt * QK + c0] = r0acc;
  out_refined[(size_t)pt * QK + c1] = r1acc;
}

// ---------------------------------------------------------------------------
extern "C" void kernel_launch(void* const* d_in, const int* in_sizes, int n_in,
                              void* d_out, int out_size, void* d_ws, size_t ws_size,
                              hipStream_t stream) {
  const float* xyz   = (const float*)d_in[0];
  const float* jafar = (const float*)d_in[1];
  const float* sem   = (const float*)d_in[2];
  const int*   knn   = (const int*)  d_in[3];
  const float* ge_w1 = (const float*)d_in[4];   const float* ge_b1 = (const float*)d_in[5];
  const float* ge_g1w= (const float*)d_in[6];   const float* ge_g1b= (const float*)d_in[7];
  const float* ge_w2 = (const float*)d_in[8];   const float* ge_b2 = (const float*)d_in[9];
  const float* ge_g2w= (const float*)d_in[10];  const float* ge_g2b= (const float*)d_in[11];
  const float* vp_w  = (const float*)d_in[12];  const float* vp_b  = (const float*)d_in[13];
  const float* vp_gw = (const float*)d_in[14];  const float* vp_gb = (const float*)d_in[15];
  const float* q_w   = (const float*)d_in[16];  const float* q_b   = (const float*)d_in[17];
  const float* k_w   = (const float*)d_in[18];  const float* k_b   = (const float*)d_in[19];
  const float* rp_w1 = (const float*)d_in[20];  const float* rp_b1 = (const float*)d_in[21];
  const float* rp_gw = (const float*)d_in[22];  const float* rp_gb = (const float*)d_in[23];
  const float* rp_w2 = (const float*)d_in[24];  const float* rp_b2 = (const float*)d_in[25];
  const float* bd_w1 = (const float*)d_in[26];  const float* bd_b1 = (const float*)d_in[27];
  const float* bd_gw = (const float*)d_in[28];  const float* bd_gb = (const float*)d_in[29];
  const float* bd_w2 = (const float*)d_in[30];  const float* bd_b2 = (const float*)d_in[31];
  const float* cls_w = (const float*)d_in[32];  const float* cls_b = (const float*)d_in[33];

  float* out = (float*)d_out;
  // Output regions (floats): logits | affinity | knn(int bits) | refined | bdy
  float* o_logits  = out;                       // [BN,13]
  float* o_aff     = out + 1703936;             // [B,N,16]
  int*   o_knn     = (int*)(out + 3801088);     // [B,N,16]
  float* o_refined = out + 5898240;             // [BN,64]
  float* o_bdy     = out + 14286848;            // [B,N]

  // Workspace: 4 big [BN,64] buffers + small stats area (~134.3 MB total).
  const size_t BIG = (size_t)BN * QK * sizeof(float);     // 33.55 MB
  char* wsb = (char*)d_ws;
  float* bufA = (float*)(wsb + 0 * BIG);    // ge1 -> g1 -> Q
  float* bufB = (float*)(wsb + 1 * BIG);    // g   -> Qw
  float* bufC = (float*)(wsb + 2 * BIG);    // bdy-hidden -> K
  float* bufD = (float*)(wsb + 3 * BIG);    // padded jafar -> V
  char*  sm   = wsb + 4 * BIG;
  float* W2T   = (float*)(sm);                 // 64*64 f32
  float* gnst  = (float*)(sm + 16384);         // <= B*8*2 f32 (reused)
  float* mom   = (float*)(sm + 16384 + 1024);  // B*9 f32
  float* rpst  = (float*)(sm + 16384 + 2048);  // B*8*2 f32
  float* clsWp = (float*)(sm + 16384 + 4096);  // padded cls_w [16,64]

  const dim3 gGemm(BN / 64), bGemm(128);
  const int eApply64 = (BN * QK + 255) / 256;
  const int eApply32 = (BN * 32 + 255) / 256;

  // Pad irregular operands (bufD is free until the V projection).
  pad_jafar<<<(BN * 32 + 255) / 256, 256, 0, stream>>>(jafar, bufD);
  pad_clsw<<<4, 256, 0, stream>>>(cls_w, clsWp);

  // geom_encoder layer 1: [BN,32(pad)] -> [BN,64], GN(8)+ReLU
  gemm_t<32, 64, 64><<<gGemm, bGemm, 0, stream>>>(bufD, ge_w1, ge_b1, bufA, BN);
  gn_stats<<<BB * 8, 256, 0, stream>>>(bufA, gnst, NN, QK, 8);
  gn_apply<<<eApply64, 256, 0, stream>>>(bufA, gnst, ge_g1w, ge_g1b, BN * QK, QK, 8, NN, 1);

  // geom_encoder layer 2: -> g in bufB
  gemm_t<64, 64, 64><<<gGemm, bGemm, 0, stream>>>(bufA, ge_w2, ge_b2, bufB, BN);
  gn_stats<<<BB * 8, 256, 0, stream>>>(bufB, gnst, NN, QK, 8);
  gn_apply<<<eApply64, 256, 0, stream>>>(bufB, gnst, ge_g2w, ge_g2b, BN * QK, QK, 8, NN, 1);

  // boundary head: g -> 32 (GN(4)+ReLU) -> 1
  gemm_t<64, 32, 32><<<gGemm, bGemm, 0, stream>>>(bufB, bd_w1, bd_b1, bufC, BN);
  gn_stats<<<BB * 4, 256, 0, stream>>>(bufC, gnst, NN, 32, 4);
  gn_apply<<<eApply32, 256, 0, stream>>>(bufC, gnst, bd_gw, bd_gb, BN * 32, 32, 4, NN, 1);
  bdy_head2<<<BN / 256, 256, 0, stream>>>(bufC, bd_w2, bd_b2, o_bdy);

  // Q (overwrites g1 in bufA), K (overwrites bdy hidden in bufC)
  gemm_t<64, 64, 64><<<gGemm, bGemm, 0, stream>>>(bufB, q_w, q_b, bufA, BN);
  gemm_t<64, 64, 64><<<gGemm, bGemm, 0, stream>>>(bufB, k_w, k_b, bufC, BN);

  // V: sem [BN,192] -> [BN,64], GN(8)+ReLU (overwrites padded jafar in bufD)
  gemm_t<192, 64, 64><<<gGemm, bGemm, 0, stream>>>(sem, vp_w, vp_b, bufD, BN);
  gn_stats<<<BB * 8, 256, 0, stream>>>(bufD, gnst, NN, QK, 8);
  gn_apply<<<eApply64, 256, 0, stream>>>(bufD, gnst, vp_gw, vp_gb, BN * QK, QK, 8, NN, 1);

  // Qw = Q @ W2  (so Q.pos_enc = Qw.h + Q.b2); overwrites g in bufB
  transpose64<<<16, 256, 0, stream>>>(rp_w2, W2T);
  gemm_t<64, 64, 64><<<gGemm, bGemm, 0, stream>>>(bufA, W2T, nullptr, bufB, BN);

  // Analytic GN stats for pos_enc hidden layer
  zero_f32<<<1, 64, 0, stream>>>(mom, BB * 9);
  rel_moments<<<dim3(256, BB), 256, 0, stream>>>(xyz, knn, mom);
  rp_finalize<<<1, 64, 0, stream>>>(mom, rp_w1, rp_b1, rpst);

  // Fused attention: one wave32 per point (8 points / 256-thread block)
  attn_fused<<<BN / 8, 256, 0, stream>>>(bufA, bufB, bufC, bufD, xyz, knn,
                                         rp_w1, rp_b1, rp_gw, rp_gb, rpst, rp_b2,
                                         o_aff, o_refined, o_knn);

  // classification head: refined [BN,64] -> logits [BN,13] (padded W, store-guarded)
  gemm_t<64, 16, NC><<<gGemm, bGemm, 0, stream>>>(o_refined, clsWp, cls_b, o_logits, BN);
}
